// RGATEncoder_74964359184928
// MI455X (gfx1250) — matrix operand order repaired and checked by reading, not compiled, not run on hardware
//
#include <hip/hip_runtime.h>
#include <hip/hip_bf16.h>

// ---------------------------------------------------------------------------
// RGAT encoder for MI455X (gfx1250, wave32, WMMA).
//   N=50000 nodes, E=800000 edges, R=8 relations, IN=H=128, B=10 graphs.
// Pipeline per layer:
//   [WMMA GEMM] xW[r] = X @ W[r]  (bf16 inputs, f32 accum), fused s_q/s_k dots
//   [edge]      alpha = leakyrelu(s_q[et,tgt] + s_k[et,src]); segment-max
//   [edge]      a = exp(alpha - m[tgt]); segment-sum den
//   [edge]      out[tgt] += (a/den[tgt]) * xW[et,src,:]   (f32 global atomics)
//   [node]      h = relu(out + bias)
// Final: out = h @ Wl^T + bl  (WMMA, transposed-B path, bias fused).
// Roofline: GEMMs ~28 GFLOP (tiny under bf16 WMMA); edge phase streams
// ~1.6 GB/layer -> memory bound at 23.3 TB/s. Weights/x/h fit in 192MB L2.
// ---------------------------------------------------------------------------

typedef __attribute__((ext_vector_type(16))) __bf16 v16bf;
typedef __attribute__((ext_vector_type(4)))  __bf16 v4bf;
typedef __attribute__((ext_vector_type(8)))  float  v8f;

#define HD 128
#define LDS_STRIDE 136   // 128 + 8 halves of padding -> bank rotation

// monotonic float -> uint key so unsigned atomicMax == float max (mixed signs ok)
__device__ __forceinline__ unsigned orderKey(float f) {
    unsigned u = __float_as_uint(f);
    return (u & 0x80000000u) ? ~u : (u | 0x80000000u);
}
__device__ __forceinline__ float keyToFloat(unsigned k) {
    unsigned u = (k & 0x80000000u) ? (k ^ 0x80000000u) : ~k;
    return __uint_as_float(u);
}

// ---------------------------------------------------------------------------
// WMMA GEMM: C[r] = A @ B_r for r in [0,R), A:[M,128] f32, W:[R,128,128] f32.
// TRANSB: B_r[k][col] = W[col*128+k] (for h @ Wl^T).
// QK:     also emit sq[r,row] = C[r,row,:].q, sk likewise (fused epilogue).
// BIAS:   C += bias[col] (final linear).
// Block: 256 thr (8 waves); wave w owns rows [m0, m0+16), all 128 cols.
// ---------------------------------------------------------------------------
template <bool TRANSB, bool QK, bool BIAS>
__global__ __launch_bounds__(256)
void wmma_gemm(const float* __restrict__ A, const float* __restrict__ W,
               const float* __restrict__ qv, const float* __restrict__ kv,
               const float* __restrict__ bias, float* __restrict__ C,
               float* __restrict__ sq, float* __restrict__ sk, int M, int R)
{
    __shared__ __attribute__((aligned(16))) __bf16 wt[HD * LDS_STRIDE];

    const int tid   = threadIdx.x;
    const int lane  = tid & 31;
    const int wave  = tid >> 5;
    const int rowl  = lane & 15;
    const int half  = lane >> 4;            // 0: lanes 0-15, 1: lanes 16-31
    const int m0    = blockIdx.x * 128 + wave * 16;
    const bool waveFull = (m0 + 16 <= M);   // wave-uniform: no exec juggling

    // ---- A fragments (row per lane, contiguous K: half h holds K 16h..16h+15), loaded once, reused for all r.
    int arow = m0 + rowl; if (arow > M - 1) arow = M - 1;
    const float* arowp = A + (size_t)arow * HD;
    v16bf afrag[4];
#pragma unroll
    for (int kk = 0; kk < 4; ++kk) {
        const float4* p4 = (const float4*)(arowp + kk * 32 + half * 16);
        float4 f0 = p4[0], f1 = p4[1];      // K = kk*32 + half*16 + (0..7)
        float4 f2 = p4[2], f3 = p4[3];      // K = kk*32 + half*16 + 8 + (0..7)
        afrag[kk][0]  = (__bf16)f0.x; afrag[kk][1]  = (__bf16)f0.y;
        afrag[kk][2]  = (__bf16)f0.z; afrag[kk][3]  = (__bf16)f0.w;
        afrag[kk][4]  = (__bf16)f1.x; afrag[kk][5]  = (__bf16)f1.y;
        afrag[kk][6]  = (__bf16)f1.z; afrag[kk][7]  = (__bf16)f1.w;
        afrag[kk][8]  = (__bf16)f2.x; afrag[kk][9]  = (__bf16)f2.y;
        afrag[kk][10] = (__bf16)f2.z; afrag[kk][11] = (__bf16)f2.w;
        afrag[kk][12] = (__bf16)f3.x; afrag[kk][13] = (__bf16)f3.y;
        afrag[kk][14] = (__bf16)f3.z; afrag[kk][15] = (__bf16)f3.w;
    }

    for (int r = 0; r < R; ++r) {
        // ---- stage B_r transposed in LDS: wt[col*LDS_STRIDE + k] (bf16)
        // 4 consecutive K per thread -> one 8B LDS store; global loads stay
        // coalesced (consecutive lanes -> consecutive cols in !TRANSB mode).
        const float* Wr = W + (size_t)r * HD * HD;
#pragma unroll 4
        for (int i = 0; i < 16; ++i) {
            int p   = tid + i * 256;          // 0..4095
            int col = p & 127;
            int k4  = (p >> 7) * 4;           // 0,4,...,124
            float w0, w1, w2, w3;
            if (TRANSB) {
                float4 f = *(const float4*)(Wr + (size_t)col * HD + k4);
                w0 = f.x; w1 = f.y; w2 = f.z; w3 = f.w;
            } else {
                w0 = Wr[(size_t)(k4 + 0) * HD + col];
                w1 = Wr[(size_t)(k4 + 1) * HD + col];
                w2 = Wr[(size_t)(k4 + 2) * HD + col];
                w3 = Wr[(size_t)(k4 + 3) * HD + col];
            }
            v4bf pk = { (__bf16)w0, (__bf16)w1, (__bf16)w2, (__bf16)w3 };
            *(v4bf*)&wt[col * LDS_STRIDE + k4] = pk;   // 8B aligned
        }
        __syncthreads();

        float dq[8], dk[8];
        if (QK) {
#pragma unroll
            for (int i = 0; i < 8; ++i) { dq[i] = 0.f; dk[i] = 0.f; }
        }

#pragma unroll
        for (int nb = 0; nb < 8; ++nb) {
            const int colg = nb * 16 + rowl;
            v8f acc = {};
#pragma unroll
            for (int kk = 0; kk < 4; ++kk) {
                // B fragment: 16 consecutive K halves at fixed col (32B, 16B-aligned)
                const __bf16* bp = &wt[colg * LDS_STRIDE + kk * 32 + half * 16];
                v16bf bfrag = *(const v16bf*)bp;
                acc = __builtin_amdgcn_wmma_f32_16x16x32_bf16(
                        false, afrag[kk], false, bfrag, (short)0, acc, false, false);
            }
            if (QK) {
                const float qc = qv[colg], kc = kv[colg];
#pragma unroll
                for (int i = 0; i < 8; ++i) { dq[i] += acc[i] * qc; dk[i] += acc[i] * kc; }
            }
            if (BIAS) {
                const float bv = bias[colg];
#pragma unroll
                for (int i = 0; i < 8; ++i) acc[i] += bv;
            }
            float* crow = C + ((size_t)r * M + m0 + half * 8) * HD + colg;
            if (waveFull) {                 // uniform branch: plain stores
#pragma unroll
                for (int i = 0; i < 8; ++i) crow[(size_t)i * HD] = acc[i];
            } else {
#pragma unroll
                for (int i = 0; i < 8; ++i)
                    if (m0 + half * 8 + i < M) crow[(size_t)i * HD] = acc[i];
            }
        }

        if (QK) {
            // reduce across the 16 lanes sharing a row (xor 1,2,4,8 stays in half)
#pragma unroll
            for (int i = 0; i < 8; ++i) {
#pragma unroll
                for (int off = 1; off < 16; off <<= 1) {
                    dq[i] += __shfl_xor(dq[i], off, 32);
                    dk[i] += __shfl_xor(dk[i], off, 32);
                }
            }
            if (rowl == 0) {
                const size_t rb = (size_t)r * M + m0 + half * 8;
                if (waveFull) {
#pragma unroll
                    for (int i = 0; i < 8; ++i) { sq[rb + i] = dq[i]; sk[rb + i] = dk[i]; }
                } else {
#pragma unroll
                    for (int i = 0; i < 8; ++i)
                        if (m0 + half * 8 + i < M) { sq[rb + i] = dq[i]; sk[rb + i] = dk[i]; }
                }
            }
        }
        __syncthreads();
    }
}

// ---------------------------------------------------------------------------
__global__ void init_buffers(unsigned* __restrict__ mkey, float* __restrict__ den,
                             float* __restrict__ agg, int N)
{
    int idx = blockIdx.x * 256 + threadIdx.x;
    if (idx < N * HD) agg[idx] = 0.f;
    if (idx < N) { mkey[idx] = 0u; den[idx] = 0.f; }   // key 0 < key(any float)
}

__global__ void edge_alpha_max(const int* __restrict__ src, const int* __restrict__ tgt,
                               const int* __restrict__ et,  const float* __restrict__ sq,
                               const float* __restrict__ sk, float* __restrict__ alpha,
                               unsigned* __restrict__ mkey, int E, int N)
{
    int e = blockIdx.x * 256 + threadIdx.x;
    if (e >= E) return;
    int s = src[e], t = tgt[e], r = et[e];
    float al = sq[(size_t)r * N + t] + sk[(size_t)r * N + s];
    al = al > 0.f ? al : 0.2f * al;                    // leaky_relu(0.2)
    alpha[e] = al;
    atomicMax(&mkey[t], orderKey(al));                 // global_atomic_max_u32
}

__global__ void edge_expsum(const int* __restrict__ tgt, float* __restrict__ alpha,
                            const unsigned* __restrict__ mkey, float* __restrict__ den, int E)
{
    int e = blockIdx.x * 256 + threadIdx.x;
    if (e >= E) return;
    int t = tgt[e];
    float a = __expf(alpha[e] - keyToFloat(mkey[t]));
    alpha[e] = a;
    atomicAdd(&den[t], a);                             // global_atomic_add_f32
}

// one wave per edge: 32 lanes x float4 = 128 cols
__global__ __launch_bounds__(256)
void edge_aggregate(const int* __restrict__ src, const int* __restrict__ tgt,
                    const int* __restrict__ et,  const float* __restrict__ alpha,
                    const float* __restrict__ den, const float* __restrict__ xW,
                    float* __restrict__ agg, int E, int N)
{
    int e = blockIdx.x * 8 + (threadIdx.x >> 5);
    if (e >= E) return;
    int lane = threadIdx.x & 31;
    int s = src[e], t = tgt[e], r = et[e];
    float coef = alpha[e] / (den[t] + 1e-16f);
    const float4* row = (const float4*)(xW + ((size_t)r * N + s) * HD);
    float* outrow = agg + (size_t)t * HD;
    float4 v = row[lane];
    atomicAdd(outrow + lane * 4 + 0, coef * v.x);
    atomicAdd(outrow + lane * 4 + 1, coef * v.y);
    atomicAdd(outrow + lane * 4 + 2, coef * v.z);
    atomicAdd(outrow + lane * 4 + 3, coef * v.w);
}

__global__ void bias_relu(const float* __restrict__ agg, const float* __restrict__ b,
                          float* __restrict__ h, int NH)
{
    int idx = blockIdx.x * 256 + threadIdx.x;
    if (idx >= NH) return;
    float v = agg[idx] + b[idx & (HD - 1)];
    h[idx] = v > 0.f ? v : 0.f;
}

// ---------------------------------------------------------------------------
extern "C" void kernel_launch(void* const* d_in, const int* in_sizes, int n_in,
                              void* d_out, int out_size, void* d_ws, size_t ws_size,
                              hipStream_t stream)
{
    const float* x    = (const float*)d_in[0];
    const int*   eidx = (const int*)  d_in[1];
    const int*   et   = (const int*)  d_in[2];
    const float* W1   = (const float*)d_in[4];
    const float* q1   = (const float*)d_in[5];
    const float* k1   = (const float*)d_in[6];
    const float* b1   = (const float*)d_in[7];
    const float* W2   = (const float*)d_in[8];
    const float* q2   = (const float*)d_in[9];
    const float* k2   = (const float*)d_in[10];
    const float* b2   = (const float*)d_in[11];
    const float* Wl   = (const float*)d_in[12];
    const float* bl   = (const float*)d_in[13];

    const int N = in_sizes[0] / HD;           // 50000
    const int E = in_sizes[1] / 2;            // 800000
    const int R = in_sizes[4] / (HD * HD);    // 8
    const int* src = eidx;
    const int* tgt = eidx + E;

    // workspace carve-out (256B aligned)
    char* base = (char*)d_ws;
    size_t off = 0;
    auto carve = [&](size_t bytes) { char* p = base + off; off += (bytes + 255) & ~(size_t)255; return p; };
    float*    xW    = (float*)   carve((size_t)R * N * HD * sizeof(float)); // 204.8 MB
    float*    h     = (float*)   carve((size_t)N * HD * sizeof(float));     // 25.6 MB
    float*    agg   = (float*)   carve((size_t)N * HD * sizeof(float));     // 25.6 MB
    float*    sq    = (float*)   carve((size_t)R * N * sizeof(float));
    float*    sk    = (float*)   carve((size_t)R * N * sizeof(float));
    float*    alpha = (float*)   carve((size_t)E * sizeof(float));
    unsigned* mkey  = (unsigned*)carve((size_t)N * sizeof(unsigned));
    float*    den   = (float*)   carve((size_t)N * sizeof(float));

    const int gemmBlocks = (N + 127) / 128;
    const int nodeBlocks = (N * HD + 255) / 256;
    const int edgeBlocks = (E + 255) / 256;
    const int aggBlocks  = (E + 7) / 8;

    // ---------------- layer 1 ----------------
    wmma_gemm<false, true, false><<<gemmBlocks, 256, 0, stream>>>(
        x, W1, q1, k1, nullptr, xW, sq, sk, N, R);
    init_buffers<<<nodeBlocks, 256, 0, stream>>>(mkey, den, agg, N);
    edge_alpha_max<<<edgeBlocks, 256, 0, stream>>>(src, tgt, et, sq, sk, alpha, mkey, E, N);
    edge_expsum<<<edgeBlocks, 256, 0, stream>>>(tgt, alpha, mkey, den, E);
    edge_aggregate<<<aggBlocks, 256, 0, stream>>>(src, tgt, et, alpha, den, xW, agg, E, N);
    bias_relu<<<nodeBlocks, 256, 0, stream>>>(agg, b1, h, N * HD);

    // ---------------- layer 2 ----------------
    wmma_gemm<false, true, false><<<gemmBlocks, 256, 0, stream>>>(
        h, W2, q2, k2, nullptr, xW, sq, sk, N, R);
    init_buffers<<<nodeBlocks, 256, 0, stream>>>(mkey, den, agg, N);
    edge_alpha_max<<<edgeBlocks, 256, 0, stream>>>(src, tgt, et, sq, sk, alpha, mkey, E, N);
    edge_expsum<<<edgeBlocks, 256, 0, stream>>>(tgt, alpha, mkey, den, E);
    edge_aggregate<<<aggBlocks, 256, 0, stream>>>(src, tgt, et, alpha, den, xW, agg, E, N);
    bias_relu<<<nodeBlocks, 256, 0, stream>>>(agg, b2, h, N * HD);

    // ---------------- final linear: out = h @ Wl^T + bl ----------------
    wmma_gemm<true, false, true><<<gemmBlocks, 256, 0, stream>>>(
        h, Wl, nullptr, nullptr, bl, (float*)d_out, nullptr, nullptr, N, 1);
}
